// PanAttBlock_2482491097152
// MI455X (gfx1250) — matrix operand order repaired and checked by reading, not compiled, not run on hardware
//
#include <hip/hip_runtime.h>
#include <hip/hip_bf16.h>
#include <cstdint>

typedef __attribute__((ext_vector_type(16))) _Float16 v16h;
typedef __attribute__((ext_vector_type(8)))  _Float16 v8h;
typedef __attribute__((ext_vector_type(8)))  float    v8f;
typedef int v4i __attribute__((vector_size(16)));

#define DEVINL __device__ __forceinline__
#define GLOBAL_AS __attribute__((address_space(1)))
#define LDS_AS    __attribute__((address_space(3)))

#if defined(__has_builtin)
# if __has_builtin(__builtin_amdgcn_global_load_async_to_lds_b128)
#  define ASYNC_MODE 1
# endif
#endif
#ifndef ASYNC_MODE
# define ASYNC_MODE 2
# warning "async-to-LDS builtin not found: falling back to inline asm global_load_async_to_lds_b128"
#endif

// Copy 16 aligned bytes global -> LDS via the CDNA5 async path (ASYNCcnt-tracked).
DEVINL void cp16(_Float16* dst_lds, const _Float16* src_glb){
#if ASYNC_MODE == 1
  __builtin_amdgcn_global_load_async_to_lds_b128(
      (GLOBAL_AS v4i*)(uintptr_t)src_glb,
      (LDS_AS v4i*)(uintptr_t)(unsigned)(uintptr_t)dst_lds, 0, 0);
#else
  unsigned l = (unsigned)(uintptr_t)dst_lds;
  asm volatile("global_load_async_to_lds_b128 %0, %1, off"
               :: "v"(l), "v"((unsigned long long)(uintptr_t)src_glb)
               : "memory");
#endif
}

#if defined(__has_builtin)
# if __has_builtin(__builtin_amdgcn_s_wait_asynccnt)
#  define WAIT_ASYNC(n) __builtin_amdgcn_s_wait_asynccnt(n)
# else
#  define WAIT_ASYNC(n) asm volatile("s_wait_asynccnt " #n ::: "memory")
# endif
#else
# define WAIT_ASYNC(n) asm volatile("s_wait_asynccnt " #n ::: "memory")
#endif

DEVINL void cp_wait_le0(){ WAIT_ASYNC(0); }
DEVINL void cp_wait_le2(){ WAIT_ASYNC(2); }
DEVINL void cp_wait_le4(){ WAIT_ASYNC(4); }
DEVINL void cp_wait(){ cp_wait_le0(); }

DEVINL int pix_n(int p){ int y=p>>8, x=p&255; return ((y>>5)<<3)+(x>>5); }
DEVINL int pix_l(int p){ int y=p>>8, x=p&255; return ((y&31)<<5)+(x&31); }
DEVINL int patch_pix(int n, int l){ int py=n>>3, px=n&7; int ly=l>>5, lx=l&31;
                                    return ((py*32+ly)<<8)+(px*32+lx); }

// A fragment: 16(M) x 32(K) f16 row-major, ld = row stride.
// lanes 0-15 -> M=lane, K=0..15; lanes 16-31 -> K=16..31.
DEVINL v16h frag_A(const _Float16* a, int ld){
  int lane = threadIdx.x & 31;
  const _Float16* r = a + (lane & 15)*ld + ((lane >> 4) << 4);
  v16h f;
#pragma unroll
  for (int t=0;t<16;++t) f[t] = r[t];
  return f;
}

// B fragment from B^T stored 16(N) x 32(K) row-major.
// lanes 0-15 -> N=lane, K=0..15; lanes 16-31 -> N=lane-16, K=16..31.
DEVINL v16h frag_B(const _Float16* bt, int ld){
  int lane = threadIdx.x & 31;
  const _Float16* r = bt + (lane & 15)*ld + ((lane >> 4) << 4);
  v16h f;
#pragma unroll
  for (int t=0;t<16;++t) f[t] = r[t];
  return f;
}

DEVINL v8f wmma16(v16h a, v16h b, v8f c){
  return __builtin_amdgcn_wmma_f32_16x16x32_f16(false, a, false, b, (short)0, c, false, false);
}

// ---------------- K0: pack pan -> NHWC f16, ms -> patch [n][c][l] f16 ---------------
__global__ void __launch_bounds__(256) k_pack(const float* __restrict__ ms,
                                              const float* __restrict__ pan,
                                              _Float16* __restrict__ pan_h,
                                              _Float16* __restrict__ msp){
  __shared__ __align__(16) float T[64*64];
  int tid = threadIdx.x;
  int pb = blockIdx.x*64;
  for (int e=tid; e<4096; e+=256){
    int c = e>>6, j = e&63;
    T[c*64 + j] = pan[(size_t)c*65536 + pb + j];
    int p = pb + j;
    msp[((size_t)pix_n(p)*64 + c)*1024 + pix_l(p)] = (_Float16)ms[(size_t)c*65536 + p];
  }
  __syncthreads();
  for (int e=tid; e<4096; e+=256){
    int j = e>>6, c = e&63;
    pan_h[(size_t)(pb+j)*64 + c] = (_Float16)T[c*64 + j];
  }
}

// ---------------- K1: implicit-GEMM KSxKS conv over NHWC f16 input ------------------
// Tile: 64 oc x 64 contiguous pixels. 8 waves, 2 C-tiles/wave. Bt staged via async.
// (No async double-buffering here: border blocks have divergent OOB zero-fill, which
//  makes per-wave ASYNCcnt arithmetic inexact, so we wait-to-zero each (kh,kw) step.)
template<int KS, int CIN, int COUT, bool PATCH_OUT, bool RELU>
__global__ void __launch_bounds__(256)
k_conv(const _Float16* __restrict__ in, const float* __restrict__ w,
       const float* __restrict__ bias, _Float16* __restrict__ out){
  constexpr int PAD = KS/2;
  constexpr int CH  = CIN/8;            // 16B chunks per pixel row
  __shared__ __align__(16) _Float16 Ws[64*CIN];   // [oc][ic]
  __shared__ __align__(16) _Float16 Bt[64*CIN];   // [px][ic]  (B^T)
  int tid = threadIdx.x, wid = tid>>5, lane = tid&31;
  int pb = blockIdx.x*64;
  int y0 = pb>>8, x0 = pb&255;
  int ocb = blockIdx.y*64;
  v8f a0 = {}, a1 = {};
  int id0 = 2*wid, id1 = 2*wid+1;
  int mt0 = id0&3, nt0 = id0>>2, mt1 = id1&3, nt1 = id1>>2;
  for (int kh=0; kh<KS; ++kh){
    int yy = y0 + kh - PAD;
    for (int kw=0; kw<KS; ++kw){
      __syncthreads();
      for (int e=tid; e<64*CH; e+=256){
        int nn = e / CH, ch = e - nn*CH;
        int xx = x0 + nn + kw - PAD;
        _Float16* dst = Bt + nn*CIN + ch*8;
        if ((unsigned)yy < 256u && (unsigned)xx < 256u){
          cp16(dst, in + (size_t)(yy*256+xx)*CIN + ch*8);
        } else {
          v8h z = {};
          *(v8h*)dst = z;
        }
      }
      for (int e=tid; e<64*CIN; e+=256){
        int m = e / CIN, ic = e - m*CIN;
        Ws[m*CIN+ic] = (_Float16)w[(size_t)(((ocb+m)*CIN+ic)*KS+kh)*KS + kw];
      }
      cp_wait();
      __syncthreads();
#pragma unroll
      for (int kc=0; kc<CIN/32; ++kc){
        a0 = wmma16(frag_A(Ws + (mt0*16)*CIN + kc*32, CIN),
                    frag_B(Bt + (nt0*16)*CIN + kc*32, CIN), a0);
        a1 = wmma16(frag_A(Ws + (mt1*16)*CIN + kc*32, CIN),
                    frag_B(Bt + (nt1*16)*CIN + kc*32, CIN), a1);
      }
    }
  }
  int rb = (lane>>4)*8, col = lane&15;
  auto st = [&](v8f a, int mt, int nt){
    int p = pb + nt*16 + col;
    v8h hv;
#pragma unroll
    for (int r=0;r<8;++r){
      float v = a[r] + bias[ocb + mt*16 + rb + r];
      if (RELU) v = fmaxf(v, 0.f);
      hv[r] = (_Float16)v;
    }
    if (PATCH_OUT){
      *(v8h*)(out + ((size_t)pix_n(p)*1024 + pix_l(p))*64 + mt*16 + rb) = hv;
    } else {
      *(v8h*)(out + (size_t)p*COUT + ocb + mt*16 + rb) = hv;
    }
  };
  st(a0, mt0, nt0); st(a1, mt1, nt1);
}

// ---------------- K2: fused q/k 1x1-conv GEMM; x,q,k in [n][l][c] -------------------
__global__ void __launch_bounds__(256)
k_qk(const _Float16* __restrict__ xs,
     const float* __restrict__ wq, const float* __restrict__ bq,
     const float* __restrict__ wk, const float* __restrict__ bk,
     _Float16* __restrict__ qb, _Float16* __restrict__ kbuf){
  __shared__ __align__(16) _Float16 Xt[64*64];   // [l][c]  B^T
  __shared__ __align__(16) _Float16 Wq[64*64];
  __shared__ __align__(16) _Float16 Wk[64*64];
  int b = blockIdx.x;
  int lt = b & 15; int n = (b>>4)&63; int s = b>>10;
  int l0 = lt*64;
  int tid = threadIdx.x, wid = tid>>5, lane = tid&31;
  const _Float16* xp = xs + ((size_t)(s*64+n)*1024)*64;
  for (int e=tid; e<512; e+=256)
    cp16(Xt + e*8, xp + (size_t)l0*64 + e*8);        // contiguous 8KB block
  for (int e=tid; e<4096; e+=256){ Wq[e] = (_Float16)wq[e]; Wk[e] = (_Float16)wk[e]; }
  cp_wait();
  __syncthreads();
  int o = wid>>2;                       // waves 0-3 -> q, 4-7 -> k
  const _Float16* W = o ? Wk : Wq;
  const float*    bb = o ? bk : bq;
  _Float16*       ob = o ? kbuf : qb;
  int rb = (lane>>4)*8, col = lane&15;
  for (int t=0;t<4;++t){
    int id = (wid&3)*4 + t;
    int mt = id&3, nt = id>>2;
    v8f a = {};
    a = wmma16(frag_A(W + (mt*16)*64,      64), frag_B(Xt + (nt*16)*64,      64), a);
    a = wmma16(frag_A(W + (mt*16)*64 + 32, 64), frag_B(Xt + (nt*16)*64 + 32, 64), a);
    int l = l0 + nt*16 + col;
    v8h hv;
#pragma unroll
    for (int r=0;r<8;++r) hv[r] = (_Float16)(a[r] + bb[mt*16 + rb + r]);
    *(v8h*)(ob + ((size_t)(s*64+n)*1024 + l)*64 + mt*16 + rb) = hv;
  }
}

// ---------------- K3: softmax pass 1 — row max & sumexp of scale*q^T k --------------
// Double-buffered: DMA for j-chunk jc+1 overlaps the 16 WMMAs of chunk jc.
__global__ void __launch_bounds__(256)
k_pass1(const _Float16* __restrict__ qb, const _Float16* __restrict__ kbuf,
        float* __restrict__ rowm, float* __restrict__ rowd){
  __shared__ __align__(16) _Float16 QT[128*64];     // [i][c]
  __shared__ __align__(16) _Float16 KT[2][128*64];  // [j][c], double-buffered
  int b = blockIdx.x;
  int blk = b & 7; int n = (b>>3)&63; int s = b>>9;
  int tid = threadIdx.x, wid = tid>>5, lane = tid&31;
  const _Float16* qp = qb   + ((size_t)(s*64+n)*1024)*64;
  const _Float16* kp = kbuf + ((size_t)(s*64+n)*1024)*64;
  for (int e=tid; e<1024; e+=256)
    cp16(QT + e*8, qp + (size_t)blk*128*64 + e*8);   // 4 async/wave
  for (int e=tid; e<1024; e+=256)
    cp16(KT[0] + e*8, kp + e*8);                     // 4 async/wave
  float m_run[8], d_run[8];
#pragma unroll
  for (int r=0;r<8;++r){ m_run[r] = -1e30f; d_run[r] = 0.f; }
  const float scale = 0.125f;
  for (int jc=0; jc<8; ++jc){
    int cur = jc & 1;
    __syncthreads();                     // everyone done reading KT[1-cur]
    if (jc < 7){
      const _Float16* src = kp + (size_t)(jc+1)*128*64;
      _Float16* dst = KT[1-cur];
      for (int e=tid; e<1024; e+=256) cp16(dst + e*8, src + e*8);
      cp_wait_le4();                     // newest 4 = prefetch; all older are done
    } else {
      cp_wait_le0();
    }
    __syncthreads();
    const _Float16* KTc = KT[cur];
    for (int jt=0; jt<8; ++jt){
      v8f sacc = {};
      sacc = wmma16(frag_A(QT + (wid*16)*64,      64), frag_B(KTc + (jt*16)*64,      64), sacc);
      sacc = wmma16(frag_A(QT + (wid*16)*64 + 32, 64), frag_B(KTc + (jt*16)*64 + 32, 64), sacc);
#pragma unroll
      for (int r=0;r<8;++r){
        float sv = sacc[r]*scale;
        float mt = sv;
#pragma unroll
        for (int off=1; off<16; off<<=1) mt = fmaxf(mt, __shfl_xor(mt, off, 32));
        float mn = fmaxf(m_run[r], mt);
        float e1 = __expf(sv - mn);
#pragma unroll
        for (int off=1; off<16; off<<=1) e1 += __shfl_xor(e1, off, 32);
        d_run[r] = d_run[r]*__expf(m_run[r]-mn) + e1;
        m_run[r] = mn;
      }
    }
  }
  if ((lane & 15) == 0){
    int base = (s*64+n)*1024 + blk*128 + wid*16 + (lane>>4)*8;
#pragma unroll
    for (int r=0;r<8;++r){ rowm[base+r] = m_run[r]; rowd[base+r] = d_run[r]; }
  }
}

// ---------------- K4: pass 2 — recompute S, P=exp/d, out = msp @ P ------------------
// QT/MS double-buffered: next i-chunk's DMA overlaps this chunk's 4 WMMAs + exp.
__global__ void __launch_bounds__(256)
k_pass2(const _Float16* __restrict__ qb, const _Float16* __restrict__ kbuf,
        const _Float16* __restrict__ msp,
        const float* __restrict__ rowm, const float* __restrict__ rowd,
        _Float16* __restrict__ catp){
  __shared__ __align__(16) _Float16 KTj[64*64];     // [j][c]
  __shared__ __align__(16) _Float16 QT[2][32*64];   // [i][c]
  __shared__ __align__(16) _Float16 MS[2][64*32];   // [c][i]  (A for out-GEMM)
  __shared__ __align__(16) _Float16 PT[64*32];      // [j][i]  (B^T for out-GEMM)
  int b = blockIdx.x;
  int jt_ = b & 15; int n = (b>>4)&63; int s = b>>10;
  int j0 = jt_*64;
  int tid = threadIdx.x, wid = tid>>5, lane = tid&31;
  const _Float16* qp = qb   + ((size_t)(s*64+n)*1024)*64;
  const _Float16* kp = kbuf + ((size_t)(s*64+n)*1024)*64;
  const _Float16* mp = msp  + (size_t)n*64*1024;
  int snb = (s*64+n)*1024;
  for (int e=tid; e<512; e+=256)
    cp16(KTj + e*8, kp + (size_t)j0*64 + e*8);       // 2 async/wave
  {
    int e = tid;
    cp16(QT[0] + e*8, qp + e*8);                     // 1 async/wave
    int c = e>>2, ch = e&3;
    cp16(MS[0] + c*32 + ch*8, mp + (size_t)c*1024 + ch*8);  // 1 async/wave
  }
  v8f out0 = {}, out1 = {};
  int id0 = 2*wid, id1 = 2*wid+1;
  int it = wid & 1, jtl = wid >> 1;
  const float scale = 0.125f;
  int rb = (lane>>4)*8;
  int buf = 0;
  for (int i0=0; i0<1024; i0+=32, buf^=1){
    __syncthreads();                    // prior iteration's reads of both buffers done
    if (i0 + 32 < 1024){
      int nb = buf^1, inext = i0 + 32;
      int e = tid;
      cp16(QT[nb] + e*8, qp + (size_t)inext*64 + e*8);
      int c = e>>2, ch = e&3;
      cp16(MS[nb] + c*32 + ch*8, mp + (size_t)c*1024 + inext + ch*8);
      cp_wait_le2();                    // newest 2 = prefetch; current buffers ready
    } else {
      cp_wait_le0();
    }
    __syncthreads();
    v8f sacc = {};
    sacc = wmma16(frag_A(QT[buf] + (it*16)*64,      64), frag_B(KTj + (jtl*16)*64,      64), sacc);
    sacc = wmma16(frag_A(QT[buf] + (it*16)*64 + 32, 64), frag_B(KTj + (jtl*16)*64 + 32, 64), sacc);
    int jloc = jtl*16 + (lane & 15);
    v8h ph;
#pragma unroll
    for (int r=0;r<8;++r){
      int i = i0 + it*16 + rb + r;
      float m = rowm[snb + i];
      float d = rowd[snb + i];
      ph[r] = (_Float16)(__expf(sacc[r]*scale - m) / d);
    }
    *(v8h*)(PT + jloc*32 + it*16 + rb) = ph;
    __syncthreads();
    {
      int mt = id0 & 3, nt = id0 >> 2;
      out0 = wmma16(frag_A(MS[buf] + (mt*16)*32, 32), frag_B(PT + (nt*16)*32, 32), out0);
    }
    {
      int mt = id1 & 3, nt = id1 >> 2;
      out1 = wmma16(frag_A(MS[buf] + (mt*16)*32, 32), frag_B(PT + (nt*16)*32, 32), out1);
    }
  }
  auto st = [&](v8f a, int id){
    int mt = id & 3, nt = id >> 2;
    int l = j0 + nt*16 + (lane & 15);
    int p = patch_pix(n, l);
    v8h hv;
#pragma unroll
    for (int r=0;r<8;++r) hv[r] = (_Float16)a[r];
    *(v8h*)(catp + (size_t)p*192 + s*64 + mt*16 + rb) = hv;
  };
  st(out0, id0); st(out1, id1);
}

// ---------------- K6: depthwise 3x3 (192 ch), NHWC, VALU ----------------------------
__global__ void __launch_bounds__(256)
k_dw(const _Float16* __restrict__ y1, const float* __restrict__ wdw,
     const float* __restrict__ bdw, _Float16* __restrict__ y2){
  int idx = blockIdx.x*256 + threadIdx.x;
  if (idx >= 192*65536) return;
  int p = idx / 192, c = idx - p*192;
  int y = p >> 8, x = p & 255;
  float a = bdw[c];
#pragma unroll
  for (int kh=0;kh<3;++kh){
    int yy = y + kh - 1;
    if ((unsigned)yy >= 256u) continue;
#pragma unroll
    for (int kw=0;kw<3;++kw){
      int xx = x + kw - 1;
      if ((unsigned)xx >= 256u) continue;
      a += (float)y1[(size_t)(yy*256+xx)*192 + c] * wdw[c*9 + kh*3 + kw];
    }
  }
  y2[idx] = (_Float16)a;
}

// ---------------- K7: pointwise GEMM 192->64 + bias + residual ----------------------
__global__ void __launch_bounds__(256)
k_pw(const _Float16* __restrict__ y2, const float* __restrict__ wpw,
     const float* __restrict__ bpw, const float* __restrict__ pan,
     float* __restrict__ outp){
  __shared__ __align__(16) _Float16 Wp[64*192];
  __shared__ __align__(16) _Float16 Bt[64*192];
  int tid = threadIdx.x, wid = tid>>5, lane = tid&31;
  int pb = blockIdx.x*64;
  for (int e=tid; e<1536; e+=256){
    int nn = e / 24, ch = e - nn*24;
    cp16(Bt + nn*192 + ch*8, y2 + (size_t)(pb+nn)*192 + ch*8);
  }
  for (int e=tid; e<64*192; e+=256){
    int m = e / 192, ic = e - m*192;
    Wp[m*192+ic] = (_Float16)wpw[m*192+ic];
  }
  cp_wait();
  __syncthreads();
  v8f a0 = {}, a1 = {};
  int id0 = 2*wid, id1 = 2*wid+1;
  int mt0 = id0&3, nt0 = id0>>2, mt1 = id1&3, nt1 = id1>>2;
#pragma unroll
  for (int kc=0; kc<6; ++kc){
    a0 = wmma16(frag_A(Wp + (mt0*16)*192 + kc*32, 192),
                frag_B(Bt + (nt0*16)*192 + kc*32, 192), a0);
    a1 = wmma16(frag_A(Wp + (mt1*16)*192 + kc*32, 192),
                frag_B(Bt + (nt1*16)*192 + kc*32, 192), a1);
  }
  int rb = (lane>>4)*8, col = lane&15;
  auto st = [&](v8f a, int mt, int nt){
    int p = pb + nt*16 + col;
#pragma unroll
    for (int r=0;r<8;++r){
      int m = mt*16 + rb + r;
      outp[(size_t)m*65536 + p] = a[r] + bpw[m] + pan[(size_t)m*65536 + p];
    }
  };
  st(a0, mt0, nt0); st(a1, mt1, nt1);
}

extern "C" void kernel_launch(void* const* d_in, const int* in_sizes, int n_in,
                              void* d_out, int out_size, void* d_ws, size_t ws_size,
                              hipStream_t stream){
  const float* ms  = (const float*)d_in[0];
  const float* pan = (const float*)d_in[1];
  const float* w1  = (const float*)d_in[2];  const float* b1  = (const float*)d_in[3];
  const float* w3  = (const float*)d_in[4];  const float* b3  = (const float*)d_in[5];
  const float* w5  = (const float*)d_in[6];  const float* b5  = (const float*)d_in[7];
  const float* wq  = (const float*)d_in[8];  const float* bq  = (const float*)d_in[9];
  const float* wk  = (const float*)d_in[10]; const float* bk  = (const float*)d_in[11];
  const float* wa  = (const float*)d_in[12]; const float* ba  = (const float*)d_in[13];
  const float* wdw = (const float*)d_in[14]; const float* bdw = (const float*)d_in[15];
  const float* wpw = (const float*)d_in[16]; const float* bpw = (const float*)d_in[17];

  char* wsp = (char*)d_ws;
  auto take = [&](size_t bytes)->void*{
    void* p = (void*)wsp; wsp += (bytes + 255) & ~(size_t)255; return p;
  };
  _Float16* pan_h = (_Float16*)take(64ull*65536*2);      // NHWC [p][64]
  _Float16* msp   = (_Float16*)take(64ull*64*1024*2);    // [n][c][l]
  _Float16* xs    = (_Float16*)take(3ull*64*1024*64*2);  // [s][n][l][c]
  _Float16* qb    = (_Float16*)take(3ull*64*1024*64*2);  // [s][n][l][c]
  _Float16* kb    = (_Float16*)take(3ull*64*1024*64*2);  // [s][n][l][c]
  float*    rowm  = (float*)take(3ull*64*1024*4);
  float*    rowd  = (float*)take(3ull*64*1024*4);
  _Float16* catp  = (_Float16*)take(65536ull*192*2);     // NHWC [p][192]
  _Float16* y1    = (_Float16*)take(65536ull*192*2);     // NHWC
  _Float16* y2    = (_Float16*)take(65536ull*192*2);     // NHWC

  k_pack<<<1024,256,0,stream>>>(ms, pan, pan_h, msp);
  k_conv<1,64,64,true,false><<<dim3(1024,1),256,0,stream>>>(pan_h, w1, b1, xs);
  k_conv<3,64,64,true,false><<<dim3(1024,1),256,0,stream>>>(pan_h, w3, b3, xs + (size_t)64*1024*64);
  k_conv<5,64,64,true,false><<<dim3(1024,1),256,0,stream>>>(pan_h, w5, b5, xs + (size_t)2*64*1024*64);
  k_qk<<<3072,256,0,stream>>>(xs, wq, bq, wk, bk, qb, kb);
  k_pass1<<<1536,256,0,stream>>>(qb, kb, rowm, rowd);
  k_pass2<<<3072,256,0,stream>>>(qb, kb, msp, rowm, rowd, catp);
  k_conv<3,192,192,false,true><<<dim3(1024,3),256,0,stream>>>(catp, wa, ba, y1);
  k_dw<<<49152,256,0,stream>>>(y1, wdw, bdw, y2);
  k_pw<<<1024,256,0,stream>>>(y2, wpw, bpw, pan, (float*)d_out);

  (void)in_sizes; (void)n_in; (void)out_size; (void)ws_size;
}
